// SimpleNEAgent_41755672052426
// MI455X (gfx1250) — matrix operand order repaired and checked by reading, not compiled, not run on hardware
//
#include <hip/hip_runtime.h>

// Problem constants (match reference)
#define B_      8192
#define IN_     512
#define NN_     1024
#define FF_     64
#define COLS_   (IN_ + NN_)       // 1536 activation columns per batch row
#define RSTRIDE 1538              // padded LDS row stride in halves: 769 dwords == 1 mod 64
                                  // -> 16 gathering rows land in 16 distinct LDS banks
#define ROWS_PER_WG 32
#define THREADS 64

typedef __attribute__((ext_vector_type(16))) _Float16 v16h;
typedef __attribute__((ext_vector_type(8)))  float    v8f;

// Per-node metadata held in registers (double buffered across loop iterations).
struct Meta {
  int4  ib0[4];                 // 16 gather indices for K = h*16 + 0..15       (WMMA #0)
  int4  ib1[4];                 // 16 gather indices for K = 32 + h*16 + 0..15  (WMMA #1)
  uint4 w0lo, w0hi, w1lo, w1hi; // f16 weights in contiguous k order (k = 16h + s), same as the B gather
};

__device__ __forceinline__ Meta load_meta(const int* __restrict__ idxs,
                                          const _Float16* __restrict__ w16,
                                          int n, int h) {
  Meta m;
  const int4* p0 = (const int4*)(idxs + n * 64 + h * 16);        // 64B aligned
  const int4* p1 = (const int4*)(idxs + n * 64 + 32 + h * 16);
#pragma unroll
  for (int k = 0; k < 4; ++k) m.ib0[k] = p0[k];
#pragma unroll
  for (int k = 0; k < 4; ++k) m.ib1[k] = p1[k];
  // A-matrix k order (contiguous, matching B gather): lanes 0-15 hold K {0..15}, lanes 16-31 hold
  // K {16..31}. All 16 lanes of a half load identical addresses (broadcast),
  // which replicates the weight vector across all 16 M rows of A.
  const uint4* wq = (const uint4*)(w16 + n * 64);                // 8 halves per uint4
  m.w0lo = wq[0 + 2 * h];   // K =      h*16 .. h*16+7
  m.w0hi = wq[1 + 2 * h];   // K =      h*16+8 ..
  m.w1lo = wq[4 + 2 * h];   // K = 32 + h*16 ..
  m.w1hi = wq[5 + 2 * h];   // K = 32 + h*16+8 ..
  return m;
}

// Kernel 1: one-time f32 -> f16 weight conversion into scratch (128 KB, L2-resident).
__global__ void __launch_bounds__(256)
cvt_weights_kernel(const float* __restrict__ w, _Float16* __restrict__ w16, int n) {
  int i = blockIdx.x * blockDim.x + threadIdx.x;
  if (i < n) w16[i] = (_Float16)w[i];
}

// Kernel 2: the DAG scan. Each wave owns 16 batch rows (B-matrix columns);
// lane%16 = batch column, lane/16 = K-half. Whole activation rows live in LDS.
__global__ void __launch_bounds__(THREADS)
neat_scan_kernel(const float* __restrict__ x, const int* __restrict__ idxs,
                 const _Float16* __restrict__ w16, float* __restrict__ out) {
  extern __shared__ _Float16 smem[];
  const int tid    = threadIdx.x;
  const int lane   = tid & 31;
  const int wid    = tid >> 5;
  const int h      = lane >> 4;        // K-half of the wave
  const int r      = lane & 15;        // batch column within this wave's tile
  const int wgrow0 = blockIdx.x * ROWS_PER_WG;

  // Stage this block's 32 input rows (f32 -> f16) into LDS. Coalesced reads.
  for (int i = tid; i < ROWS_PER_WG * IN_; i += THREADS) {
    int rr = i >> 9;                   // IN_ == 512
    int cc = i & (IN_ - 1);
    smem[rr * RSTRIDE + cc] = (_Float16)x[(size_t)(wgrow0 + rr) * IN_ + cc];
  }
  __syncthreads();

  const unsigned rb = (unsigned)(wid * 16 + r) * RSTRIDE;  // half-element row base

  Meta cur = load_meta(idxs, w16, 0, h);
  float tlast = 0.0f;

  for (int n = 0; n < NN_; ++n) {
    // Software pipeline: issue next node's metadata loads before this node's math.
    const int np  = (n + 1 < NN_) ? n + 1 : n;
    Meta nxt = load_meta(idxs, w16, np, h);

    // Deep prefetch of the index/weight stream into the NEAR cache (WGP scope)
    // so the double-buffered loads above hit L0: locality=3 -> near prefetch.
    const int pf = (n + 8 < NN_) ? n + 8 : n;
    __builtin_prefetch((const char*)(idxs + pf * 64) + lane * 8, 0, 3);
    __builtin_prefetch((const char*)(w16  + pf * 64) + lane * 4, 0, 3);

    // Gather B matrices from LDS: lane = batch column, halves = K.
    const int* bi0 = (const int*)cur.ib0;
    const int* bi1 = (const int*)cur.ib1;
    v16h b0, b1;
#pragma unroll
    for (int j = 0; j < 16; ++j) b0[j] = smem[rb + (unsigned)bi0[j]];
#pragma unroll
    for (int j = 0; j < 16; ++j) b1[j] = smem[rb + (unsigned)bi1[j]];

    union { struct { uint4 lo, hi; } u; v16h v; } A0, A1;
    A0.u.lo = cur.w0lo; A0.u.hi = cur.w0hi;
    A1.u.lo = cur.w1lo; A1.u.hi = cur.w1hi;

    // D = A x B + C ; weights replicated over M, so D[:, r] all equal out[row r].
    v8f c = {};
    c = __builtin_amdgcn_wmma_f32_16x16x32_f16(false, A0.v, false, b0,
                                               (short)0, c, false, false);
    c = __builtin_amdgcn_wmma_f32_16x16x32_f16(false, A1.v, false, b1,
                                               (short)0, c, false, false);

    // tanh via v_exp_f32 / v_rcp_f32: tanh(v) = 1 - 2/(exp2(2v*log2e)+1)
    float v = c[0];
    float E = __builtin_amdgcn_exp2f(v * 2.885390081777927f);
    float t = 1.0f - 2.0f * __builtin_amdgcn_rcpf(E + 1.0f);

    // Append node output. Lanes l and l+16 write the same value to the same slot
    // (benign). Same-wave DS pipe is in-order -> next node's gathers see it.
    smem[rb + (unsigned)(IN_ + n)] = (_Float16)t;
    tlast = t;
    cur = nxt;
  }

  // Reference returns activs[:, -1] == node N-1's output for every batch row.
  if (h == 0) out[wgrow0 + wid * 16 + r] = tlast;
}

extern "C" void kernel_launch(void* const* d_in, const int* in_sizes, int n_in,
                              void* d_out, int out_size, void* d_ws, size_t ws_size,
                              hipStream_t stream) {
  (void)in_sizes; (void)n_in; (void)out_size; (void)ws_size;
  const float* x    = (const float*)d_in[0];
  const int*   idxs = (const int*)d_in[1];
  const float* w    = (const float*)d_in[2];
  float*       out  = (float*)d_out;
  _Float16*    w16  = (_Float16*)d_ws;       // 1024*64*2 = 128 KB scratch

  cvt_weights_kernel<<<(NN_ * FF_ + 255) / 256, 256, 0, stream>>>(w, w16, NN_ * FF_);

  const size_t shmem = (size_t)ROWS_PER_WG * RSTRIDE * sizeof(_Float16); // 98432 B
  hipFuncSetAttribute(reinterpret_cast<const void*>(neat_scan_kernel),
                      hipFuncAttributeMaxDynamicSharedMemorySize, (int)shmem);
  neat_scan_kernel<<<B_ / ROWS_PER_WG, THREADS, shmem, stream>>>(x, idxs, w16, out);
}
